// MultiHeadAttention_86560770884069
// MI455X (gfx1250) — compile-verified, matrix operand order checked by reading, not run on hardware
//
#include <hip/hip_runtime.h>

// ---------------------------------------------------------------------------
// MHA forward for MI455X (gfx1250, wave32, WMMA bf16).
//   pass 0: fp32 -> bf16 pre-conversion of activations + weights (bandwidth
//           bound, native packed cvt) so GEMM loops are pure load+WMMA.
//   pass 1: Q/K/V projections (bf16 x bf16 -> f32 acc), Q pre-scaled 1/8,
//           V stored transposed [B,H,dk,S] for contiguous P@V B-operand loads.
//   pass 2: flash attention per (b,h,32-query tile), causal, online softmax,
//           P staged via LDS to convert C-layout -> A-layout.
//   pass 3: output projection (bf16 ctx @ wo.T + bo) -> fp32 d_out.
// All matmuls: v_wmma_f32_16x16x32_bf16.
// ---------------------------------------------------------------------------

#define DM 1024
#define HH 16
#define DK 64
#define BB 2
#define SS 2048

typedef unsigned int u32;
typedef unsigned short u16;

typedef __attribute__((ext_vector_type(16))) __bf16 v16bf;
typedef __attribute__((ext_vector_type(8)))  float  v8f;

union Frag {
    v16bf v;
    u32   u[8];
    uint4 q[2];
};

// Native fp32 -> bf16 (lets the backend use packed v_cvt hardware).
__device__ __forceinline__ u16 bfh(float f) {
    __bf16 h = (__bf16)f;
    return __builtin_bit_cast(u16, h);
}
__device__ __forceinline__ u32 pk2(float lo, float hi) {
    return (u32)bfh(lo) | ((u32)bfh(hi) << 16);
}
__device__ __forceinline__ v8f vzero() {
    v8f z = {0.f, 0.f, 0.f, 0.f, 0.f, 0.f, 0.f, 0.f};
    return z;
}
__device__ __forceinline__ v8f wmma_bf16(const v16bf a, const v16bf b, v8f c) {
    // D = A(16x32 bf16) * B(32x16 bf16) + C(16x16 f32)
    return __builtin_amdgcn_wmma_f32_16x16x32_bf16(false, a, false, b,
                                                   (short)0, c, false, false);
}

// Fragment load from bf16 memory. Per ISA 16-bit layout: lane holds row
// (l&15); K runs [hl*8, hl*8+8) and [16+hl*8, +8) (hl = lane>>4) ->
// two contiguous 16B runs of one row.
__device__ __forceinline__ void ldfrag(Frag& f, const u16* __restrict__ base,
                                       int ld, int row, int col, int hl) {
    const u16* p = base + (size_t)row * ld + col + hl * 8;
    f.q[0] = *(const uint4*)(p);
    f.q[1] = *(const uint4*)(p + 16);
}

// ---------------------------------------------------------------------------
// Pass 0: elementwise fp32 -> bf16, 8 elements/thread (one b128 in, b128 out
// per pair of threads' worth; fully coalesced).
// ---------------------------------------------------------------------------
__global__ __launch_bounds__(256) void cvt_kernel(const float* __restrict__ in,
                                                  u16* __restrict__ out, int n) {
    const int i = (blockIdx.x * 256 + threadIdx.x) * 8;
    if (i >= n) return;
    float4 a = *(const float4*)(in + i);
    float4 b = *(const float4*)(in + i + 4);
    uint4 o;
    o.x = pk2(a.x, a.y);
    o.y = pk2(a.z, a.w);
    o.z = pk2(b.x, b.y);
    o.w = pk2(b.z, b.w);
    *(uint4*)(out + i) = o;
}

// ---------------------------------------------------------------------------
// Pass 1: out = (X @ W.T + bias) * scale, bf16 operands, bf16 result.
// vt_layout==0: out[((b*H+h)*S+s)*DK+dk]   (Q/K layout)
// vt_layout==1: out[((b*H+h)*DK+dk)*S+s]   (V transposed layout)
// grid (M/128, N/128), block 128 (4 waves, each a 64x64 tile).
// ---------------------------------------------------------------------------
__global__ __launch_bounds__(128) void proj_kernel(
    const u16* __restrict__ X, const u16* __restrict__ W,
    const float* __restrict__ bias, u16* __restrict__ out,
    int vt_layout, float scale) {
    const int tid = threadIdx.x;
    const int w = tid >> 5, lane = tid & 31, hl = lane >> 4, lr = lane & 15;
    const int m0 = blockIdx.x * 128 + (w >> 1) * 64;
    const int n0 = blockIdx.y * 128 + (w & 1) * 64;

    v8f acc[4][4];
#pragma unroll
    for (int i = 0; i < 4; ++i)
#pragma unroll
        for (int j = 0; j < 4; ++j) acc[i][j] = vzero();

    for (int k0 = 0; k0 < DM; k0 += 32) {
        Frag af[4], bf[4];
#pragma unroll
        for (int i = 0; i < 4; ++i) ldfrag(af[i], X, DM, m0 + i * 16 + lr, k0, hl);
#pragma unroll
        for (int j = 0; j < 4; ++j) ldfrag(bf[j], W, DM, n0 + j * 16 + lr, k0, hl);
        if (k0 + 32 < DM) {
            // pull next k-step toward the WGP while WMMAs run
            __builtin_prefetch(X + (size_t)(m0 + lane) * DM + k0 + 32, 0, 1);
            __builtin_prefetch(W + (size_t)(n0 + lane) * DM + k0 + 32, 0, 1);
        }
#pragma unroll
        for (int i = 0; i < 4; ++i)
#pragma unroll
            for (int j = 0; j < 4; ++j)
                acc[i][j] = wmma_bf16(af[i].v, bf[j].v, acc[i][j]);
    }

#pragma unroll
    for (int j = 0; j < 4; ++j) {
        const int n = n0 + j * 16 + lr;
        const float bv = bias[n];
        const int h = n >> 6, dk = n & (DK - 1);
#pragma unroll
        for (int i = 0; i < 4; ++i) {
#pragma unroll
            for (int v = 0; v < 8; ++v) {
                const int t = m0 + i * 16 + v + 8 * hl;          // token row
                const int b = t >> 11, s = t & (SS - 1);
                const float val = (acc[i][j][v] + bv) * scale;
                size_t idx = vt_layout
                    ? ((size_t)(b * HH + h) * DK + dk) * SS + s
                    : ((size_t)(b * HH + h) * SS + s) * DK + dk;
                out[idx] = bfh(val);
            }
        }
    }
}

// ---------------------------------------------------------------------------
// Pass 2: causal flash attention. One wave = 32 queries x full dk=64 for one
// (b,h). Key blocks of 64. Scores/probs in VGPRs; P routed through LDS to
// convert C-layout -> A-layout for the P@V WMMA.
// grid 256, block 256 (8 independent waves).
// ---------------------------------------------------------------------------
__global__ __launch_bounds__(256) void attn_kernel(
    const u16* __restrict__ qp, const u16* __restrict__ kp,
    const u16* __restrict__ vt, u16* __restrict__ ctxo) {
    __shared__ __align__(16) u16 plds[8][32 * 64];   // 32KB: per-wave P tile

    const int tid = threadIdx.x;
    const int w = tid >> 5, lane = tid & 31, hl = lane >> 4, lr = lane & 15;
    const int gw = blockIdx.x * 8 + w;        // 0..2047
    const int bh = gw >> 6;                   // (b*H+h), 0..31
    const int qt = gw & 63;                   // query tile of 32 rows
    const int q0 = qt * 32;

    const u16* qb = qp + (size_t)bh * SS * DK;
    const u16* kb = kp + (size_t)bh * SS * DK;
    const u16* vb = vt + (size_t)bh * DK * SS;

    // Q fragments: rows q0..q0+31, K-dim = dk (two 32-steps). Loaded once.
    Frag qf[2][2];
#pragma unroll
    for (int i = 0; i < 2; ++i)
#pragma unroll
        for (int ks = 0; ks < 2; ++ks)
            ldfrag(qf[i][ks], qb, DK, q0 + i * 16 + lr, ks * 32, hl);

    v8f ctx[2][4];
    float mrow[2][8], lsum[2][8];
#pragma unroll
    for (int i = 0; i < 2; ++i) {
#pragma unroll
        for (int jd = 0; jd < 4; ++jd) ctx[i][jd] = vzero();
#pragma unroll
        for (int v = 0; v < 8; ++v) { mrow[i][v] = -1e30f; lsum[i][v] = 0.f; }
    }

    const int nkb = (q0 >> 6) + 1;            // causal: key blocks 0..q0/64
    for (int kblk = 0; kblk < nkb; ++kblk) {
        const int kstart = kblk * 64;

        // ---- S = Q @ K^T (scale already folded into Q) ----
        v8f sf[2][4];
#pragma unroll
        for (int i = 0; i < 2; ++i)
#pragma unroll
            for (int j = 0; j < 4; ++j) sf[i][j] = vzero();
#pragma unroll
        for (int ks = 0; ks < 2; ++ks) {
            Frag kf[4];
#pragma unroll
            for (int j = 0; j < 4; ++j)
                ldfrag(kf[j], kb, DK, kstart + j * 16 + lr, ks * 32, hl);
#pragma unroll
            for (int i = 0; i < 2; ++i)
#pragma unroll
                for (int j = 0; j < 4; ++j)
                    sf[i][j] = wmma_bf16(qf[i][ks].v, kf[j].v, sf[i][j]);
        }

        // ---- causal mask on the diagonal block ----
        if (kblk == nkb - 1) {
#pragma unroll
            for (int i = 0; i < 2; ++i)
#pragma unroll
                for (int j = 0; j < 4; ++j)
#pragma unroll
                    for (int v = 0; v < 8; ++v) {
                        const int row = q0 + i * 16 + v + 8 * hl;
                        const int col = kstart + j * 16 + lr;
                        if (col > row) sf[i][j][v] = -1e30f;
                    }
        }

        // ---- online softmax (row = C-layout VGPR idx within 16-lane half) ----
#pragma unroll
        for (int i = 0; i < 2; ++i) {
#pragma unroll
            for (int v = 0; v < 8; ++v) {
                float mx = sf[i][0][v];
#pragma unroll
                for (int j = 1; j < 4; ++j) mx = fmaxf(mx, sf[i][j][v]);
#pragma unroll
                for (int off = 1; off < 16; off <<= 1)
                    mx = fmaxf(mx, __shfl_xor(mx, off, 32));
                const float mo = mrow[i][v];
                const float mn = fmaxf(mo, mx);
                const float corr = __expf(mo - mn);
                mrow[i][v] = mn;
                float rs = 0.f;
#pragma unroll
                for (int j = 0; j < 4; ++j) {
                    const float p = __expf(sf[i][j][v] - mn);
                    sf[i][j][v] = p;
                    rs += p;
                }
#pragma unroll
                for (int off = 1; off < 16; off <<= 1)
                    rs += __shfl_xor(rs, off, 32);
                lsum[i][v] = lsum[i][v] * corr + rs;
#pragma unroll
                for (int jd = 0; jd < 4; ++jd) ctx[i][jd][v] *= corr;
            }
        }

        // ---- P: C-layout -> LDS (row-major bf16 32x64) ----
#pragma unroll
        for (int i = 0; i < 2; ++i)
#pragma unroll
            for (int j = 0; j < 4; ++j)
#pragma unroll
                for (int v = 0; v < 8; ++v)
                    plds[w][(i * 16 + v + 8 * hl) * 64 + j * 16 + lr] =
                        bfh(sf[i][j][v]);
        // same-wave DS ops are in-order: no barrier needed.

        // ---- ctx += P @ V  (V^T layout: rows are dk, contiguous in s) ----
#pragma unroll
        for (int ks = 0; ks < 2; ++ks) {
            Frag pf[2];
#pragma unroll
            for (int i = 0; i < 2; ++i) {
                const u16* p = &plds[w][(i * 16 + lr) * 64 + ks * 32 + hl * 8];
                pf[i].q[0] = *(const uint4*)(p);
                pf[i].q[1] = *(const uint4*)(p + 16);
            }
            Frag vf[4];
#pragma unroll
            for (int jd = 0; jd < 4; ++jd)
                ldfrag(vf[jd], vb, SS, jd * 16 + lr, kstart + ks * 32, hl);
#pragma unroll
            for (int i = 0; i < 2; ++i)
#pragma unroll
                for (int jd = 0; jd < 4; ++jd)
                    ctx[i][jd] = wmma_bf16(pf[i].v, vf[jd].v, ctx[i][jd]);
        }
    }

    // ---- epilogue: ctx/l -> bf16 workspace in [B,S,D] (heads concatenated) ----
    const int b = bh >> 4, h = bh & (HH - 1);
#pragma unroll
    for (int i = 0; i < 2; ++i)
#pragma unroll
        for (int v = 0; v < 8; ++v) {
            const float inv = 1.f / lsum[i][v];
            const int t = b * SS + q0 + i * 16 + v + 8 * hl;
#pragma unroll
            for (int jd = 0; jd < 4; ++jd) {
                const int d = h * DK + jd * 16 + lr;
                ctxo[(size_t)t * DM + d] = bfh(ctx[i][jd][v] * inv);
            }
        }
}

// ---------------------------------------------------------------------------
// Pass 3: out = ctx(bf16) @ wo(bf16).T + bo, fp32 result.
// ---------------------------------------------------------------------------
__global__ __launch_bounds__(128) void out_kernel(
    const u16* __restrict__ A, const u16* __restrict__ W,
    const float* __restrict__ bias, float* __restrict__ out) {
    const int tid = threadIdx.x;
    const int w = tid >> 5, lane = tid & 31, hl = lane >> 4, lr = lane & 15;
    const int m0 = blockIdx.x * 128 + (w >> 1) * 64;
    const int n0 = blockIdx.y * 128 + (w & 1) * 64;

    v8f acc[4][4];
#pragma unroll
    for (int i = 0; i < 4; ++i)
#pragma unroll
        for (int j = 0; j < 4; ++j) acc[i][j] = vzero();

    for (int k0 = 0; k0 < DM; k0 += 32) {
        Frag af[4], bf[4];
#pragma unroll
        for (int i = 0; i < 4; ++i) ldfrag(af[i], A, DM, m0 + i * 16 + lr, k0, hl);
#pragma unroll
        for (int j = 0; j < 4; ++j) ldfrag(bf[j], W, DM, n0 + j * 16 + lr, k0, hl);
        if (k0 + 32 < DM) {
            __builtin_prefetch(A + (size_t)(m0 + lane) * DM + k0 + 32, 0, 1);
            __builtin_prefetch(W + (size_t)(n0 + lane) * DM + k0 + 32, 0, 1);
        }
#pragma unroll
        for (int i = 0; i < 4; ++i)
#pragma unroll
            for (int j = 0; j < 4; ++j)
                acc[i][j] = wmma_bf16(af[i].v, bf[j].v, acc[i][j]);
    }

#pragma unroll
    for (int j = 0; j < 4; ++j) {
        const int n = n0 + j * 16 + lr;
        const float bv = bias[n];
#pragma unroll
        for (int i = 0; i < 4; ++i)
#pragma unroll
            for (int v = 0; v < 8; ++v) {
                const int t = m0 + i * 16 + v + 8 * hl;
                out[(size_t)t * DM + n] = acc[i][j][v] + bv;
            }
    }
}

// ---------------------------------------------------------------------------
extern "C" void kernel_launch(void* const* d_in, const int* in_sizes, int n_in,
                              void* d_out, int out_size, void* d_ws, size_t ws_size,
                              hipStream_t stream) {
    (void)in_sizes; (void)n_in; (void)out_size; (void)ws_size;
    const float* q  = (const float*)d_in[0];
    const float* k  = (const float*)d_in[1];
    const float* v  = (const float*)d_in[2];
    // d_in[3] = causal mask; implemented analytically (col > row).
    const float* wq = (const float*)d_in[4];
    const float* bq = (const float*)d_in[5];
    const float* wk = (const float*)d_in[6];
    const float* bk = (const float*)d_in[7];
    const float* wv = (const float*)d_in[8];
    const float* bv = (const float*)d_in[9];
    const float* wo = (const float*)d_in[10];
    const float* bo = (const float*)d_in[11];
    float* out = (float*)d_out;

    // bf16 workspace (u16 elements):
    //   qp|kp|vt|ctx (4M each) + xq|xk|xv (4M each) + wq|wk|wv|wo (1M each)
    //   = 32M u16 = 64MB total.
    const size_t NE = (size_t)BB * SS * DM;      // 4,194,304
    const size_t NW = (size_t)DM * DM;           // 1,048,576
    u16* qp   = (u16*)d_ws;
    u16* kpp  = qp   + NE;
    u16* vtp  = kpp  + NE;
    u16* ctxw = vtp  + NE;
    u16* xq   = ctxw + NE;
    u16* xk   = xq   + NE;
    u16* xv   = xk   + NE;
    u16* wqb  = xv   + NE;
    u16* wkb  = wqb  + NW;
    u16* wvb  = wkb  + NW;
    u16* wob  = wvb  + NW;

    // pass 0: conversions (bandwidth bound, ~3us total at 23.3 TB/s)
    const int CB = 256;
    cvt_kernel<<<(int)(NE / 8 / CB), CB, 0, stream>>>(q,  xq,  (int)NE);
    cvt_kernel<<<(int)(NE / 8 / CB), CB, 0, stream>>>(k,  xk,  (int)NE);
    cvt_kernel<<<(int)(NE / 8 / CB), CB, 0, stream>>>(v,  xv,  (int)NE);
    cvt_kernel<<<(int)(NW / 8 / CB), CB, 0, stream>>>(wq, wqb, (int)NW);
    cvt_kernel<<<(int)(NW / 8 / CB), CB, 0, stream>>>(wk, wkb, (int)NW);
    cvt_kernel<<<(int)(NW / 8 / CB), CB, 0, stream>>>(wv, wvb, (int)NW);
    cvt_kernel<<<(int)(NW / 8 / CB), CB, 0, stream>>>(wo, wob, (int)NW);

    const dim3 g(32, 8), blk(128);
    // Q pre-scaled by 1/sqrt(dk)=0.125 so attention needs no extra scale.
    proj_kernel<<<g, blk, 0, stream>>>(xq, wqb, bq, qp, 0, 0.125f);
    proj_kernel<<<g, blk, 0, stream>>>(xk, wkb, bk, kpp, 0, 1.0f);
    proj_kernel<<<g, blk, 0, stream>>>(xv, wvb, bv, vtp, 1, 1.0f);
    attn_kernel<<<256, 256, 0, stream>>>(qp, kpp, vtp, ctxw);
    out_kernel<<<g, blk, 0, stream>>>(ctxw, wob, bo, out);
}
